// DynamicQuantConv2d_62835371540985
// MI455X (gfx1250) — 8-bit matrix operand order repaired and checked by reading, compiled, not run on hardware
//
#include <hip/hip_runtime.h>
#include <stdint.h>

typedef __attribute__((ext_vector_type(8))) int v8i;

#define HWSZ      36864
#define IMG_ELEMS 1179648
#define KTOT      288
#define KPAD      320

#define WS_GMIN   0
#define WS_GMAX   3
#define WS_WMIN   6
#define WS_WMAX   7
#define WS_MNX    8
#define WS_SX     11
#define WS_SW     14
#define WS_SJF    32
#define WS_SJROW  128
#define WS_SJCOL  320
#define WS_SJCOR  512
#define WS_BFRAG  1024

__device__ __forceinline__ unsigned fkey(float f) {
  unsigned u = __float_as_uint(f);
  return (u & 0x80000000u) ? ~u : (u | 0x80000000u);
}
__device__ __forceinline__ float funkey(unsigned k) {
  unsigned u = (k & 0x80000000u) ? (k ^ 0x80000000u) : ~k;
  return __uint_as_float(u);
}

__global__ void k_init_weight(const float* __restrict__ w, void* ws) {
  unsigned* wsu = (unsigned*)ws;
  float*    wsf = (float*)ws;
  int*      wsi = (int*)ws;
  int tid = threadIdx.x;
  if (tid < 3) { wsu[WS_GMIN + tid] = 0xFFFFFFFFu; wsu[WS_GMAX + tid] = 0u; }

  __shared__ float smn[256], smx[256];
  float mn = 3.4e38f, mx = -3.4e38f;
  for (int i = tid; i < 32 * 32 * 9; i += 256) {
    float v = w[i];
    mn = fminf(mn, v); mx = fmaxf(mx, v);
  }
  smn[tid] = mn; smx[tid] = mx;
  __syncthreads();
  for (int off = 128; off > 0; off >>= 1) {
    if (tid < off) {
      smn[tid] = fminf(smn[tid], smn[tid + off]);
      smx[tid] = fmaxf(smx[tid], smx[tid + off]);
    }
    __syncthreads();
  }
  float wmin = smn[0], wmax = smx[0];
  if (tid == 0) { wsf[WS_WMIN] = wmin; wsf[WS_WMAX] = wmax; }

  if (tid < 96) {
    int g = tid / 32, n = tid % 32;
    int bbits = (g == 0) ? 2 : ((g == 1) ? 4 : 8);
    float levels = (float)((1 << bbits) - 1);
    float sw = fmaxf((wmax - wmin) / levels, 1e-8f);
    if (n == 0) wsf[WS_SW + g] = sw;
    float inv = 1.0f / sw;
    int sF = 0, sR0 = 0, sR2 = 0, sC0 = 0, sC2 = 0;
    int s00 = 0, s02 = 0, s20 = 0, s22 = 0;
    unsigned char* bb = (unsigned char*)(wsu + WS_BFRAG);
    for (int K = 0; K < KPAD; ++K) {
      int code = 0;
      if (K < KTOT) {
        int tap = K / 32, ci = K % 32, kh = tap / 3, kw = tap % 3;
        float v = w[((n * 32 + ci) * 3 + kh) * 3 + kw];
        code = (int)rintf((v - wmin) * inv);
        sF += code;
        if (kh == 0) sR0 += code;
        if (kh == 2) sR2 += code;
        if (kw == 0) sC0 += code;
        if (kw == 2) sC2 += code;
        if (kh == 0 && kw == 0) s00 += code;
        if (kh == 0 && kw == 2) s02 += code;
        if (kh == 2 && kw == 0) s20 += code;
        if (kh == 2 && kw == 2) s22 += code;
      }
      int s = K / 64, ko = K % 64, half = ko / 32, rem = ko % 32;
      int laneHi = rem / 16, wi = rem % 16;
      int vreg = half * 4 + wi / 4, byt = wi % 4;
      int lane = laneHi * 16 + (n & 15), nt = n >> 4;
      size_t idx = (size_t)(((((g * 5 + s) * 2 + nt) * 32 + lane) * 8 + vreg)) * 4 + byt;
      bb[idx] = (unsigned char)code;
    }
    wsi[WS_SJF + g * 32 + n] = sF;
    wsi[WS_SJROW + (g * 2 + 0) * 32 + n] = sR0;
    wsi[WS_SJROW + (g * 2 + 1) * 32 + n] = sR2;
    wsi[WS_SJCOL + (g * 2 + 0) * 32 + n] = sC0;
    wsi[WS_SJCOL + (g * 2 + 1) * 32 + n] = sC2;
    wsi[WS_SJCOR + ((g * 2 + 0) * 2 + 0) * 32 + n] = s00;
    wsi[WS_SJCOR + ((g * 2 + 0) * 2 + 1) * 32 + n] = s02;
    wsi[WS_SJCOR + ((g * 2 + 1) * 2 + 0) * 32 + n] = s20;
    wsi[WS_SJCOR + ((g * 2 + 1) * 2 + 1) * 32 + n] = s22;
  }
}

__global__ void k_xstats(const float* __restrict__ x, const int* __restrict__ bits,
                         void* ws) {
  unsigned* wsu = (unsigned*)ws;
  int img = blockIdx.y, chunk = blockIdx.x, tid = threadIdx.x;
  const float4* xv = (const float4*)(x + (size_t)img * IMG_ELEMS + (size_t)chunk * 16384);
  float mn = 3.4e38f, mx = -3.4e38f;
#pragma unroll
  for (int i = 0; i < 16; ++i) {
    float4 v = xv[tid + i * 256];
    mn = fminf(mn, fminf(fminf(v.x, v.y), fminf(v.z, v.w)));
    mx = fmaxf(mx, fmaxf(fmaxf(v.x, v.y), fmaxf(v.z, v.w)));
  }
  __shared__ float smn[256], smx[256];
  smn[tid] = mn; smx[tid] = mx;
  __syncthreads();
  for (int off = 128; off > 0; off >>= 1) {
    if (tid < off) {
      smn[tid] = fminf(smn[tid], smn[tid + off]);
      smx[tid] = fmaxf(smx[tid], smx[tid + off]);
    }
    __syncthreads();
  }
  if (tid == 0) {
    int bv = bits[img];
    int g = (bv == 2) ? 0 : ((bv == 4) ? 1 : 2);
    atomicMin(&wsu[WS_GMIN + g], fkey(smn[0]));
    atomicMax(&wsu[WS_GMAX + g], fkey(smx[0]));
  }
}

__global__ void k_finalize(void* ws) {
  unsigned* wsu = (unsigned*)ws;
  float*    wsf = (float*)ws;
  int t = threadIdx.x;
  if (t < 3) {
    unsigned kmin = wsu[WS_GMIN + t], kmax = wsu[WS_GMAX + t];
    bool has = (kmin != 0xFFFFFFFFu);
    float mn = has ? funkey(kmin) : 0.0f;
    float mx = has ? funkey(kmax) : 0.0f;
    int bbits = (t == 0) ? 2 : ((t == 1) ? 4 : 8);
    float sx = fmaxf((mx - mn) / (float)((1 << bbits) - 1), 1e-8f);
    wsf[WS_MNX + t] = mn;
    wsf[WS_SX + t]  = sx;
  }
}

__global__ __launch_bounds__(256) void k_conv(const float* __restrict__ x,
                                              const float* __restrict__ bias,
                                              const int* __restrict__ bits,
                                              const void* ws,
                                              float* __restrict__ out) {
  const unsigned* wsu = (const unsigned*)ws;
  const float*    wsf = (const float*)ws;
  const int*      wsi = (const int*)ws;

  __shared__ __align__(16) unsigned char xs[6 * 194 * 32];
  __shared__ float ot[8][512];

  int tid = threadIdx.x, img = blockIdx.y, rb = blockIdx.x;
  int bv = bits[img];
  int g = (bv == 2) ? 0 : ((bv == 4) ? 1 : 2);
  float mnx = wsf[WS_MNX + g], sx = wsf[WS_SX + g];
  float sww = wsf[WS_SW + g],  mnw = wsf[WS_WMIN];
  float invsx = 1.0f / sx;
  const float* ximg = x + (size_t)img * IMG_ELEMS;

  for (int p = tid; p < 6 * 194; p += 256) {
    int lr = p / 194, cp = p % 194;
    int gr = rb * 4 - 1 + lr, gc = cp - 1;
    unsigned* dst = (unsigned*)&xs[(size_t)p * 32];
    if ((unsigned)gr < 192u && (unsigned)gc < 192u) {
      const float* src = ximg + gr * 192 + gc;
#pragma unroll
      for (int c4 = 0; c4 < 8; ++c4) {
        unsigned u = 0;
#pragma unroll
        for (int j = 0; j < 4; ++j) {
          float v = src[(size_t)(c4 * 4 + j) * HWSZ];
          int code = (int)rintf((v - mnx) * invsx);
          u |= ((unsigned)code & 0xFFu) << (8 * j);
        }
        dst[c4] = u;
      }
    } else {
#pragma unroll
      for (int c4 = 0; c4 < 8; ++c4) dst[c4] = 0u;
    }
  }

  int wave = tid >> 5, lane = tid & 31, n0 = lane & 15, laneHi = (lane >> 4) & 1;

  v8i bfr[5][2];
#pragma unroll
  for (int s = 0; s < 5; ++s) {
#pragma unroll
    for (int nt = 0; nt < 2; ++nt) {
      const unsigned* q = wsu + WS_BFRAG + (((g * 5 + s) * 2 + nt) * 32 + lane) * 8;
      v8i b;
#pragma unroll
      for (int v = 0; v < 8; ++v) b[v] = (int)q[v];
      bfr[s][nt] = b;
    }
  }
  const v8i bone_full = {0x01010101, 0x01010101, 0x01010101, 0x01010101,
                         0x01010101, 0x01010101, 0x01010101, 0x01010101};
  const v8i bone_last = {0x01010101, 0x01010101, 0x01010101, 0x01010101, 0, 0, 0, 0};

  int sjF[2], sjR0[2], sjR2[2], sjC0[2], sjC2[2], sj00[2], sj02[2], sj20[2], sj22[2];
  float biasn[2];
#pragma unroll
  for (int nt = 0; nt < 2; ++nt) {
    int n = nt * 16 + n0;
    sjF[nt]  = wsi[WS_SJF + g * 32 + n];
    sjR0[nt] = wsi[WS_SJROW + (g * 2 + 0) * 32 + n];
    sjR2[nt] = wsi[WS_SJROW + (g * 2 + 1) * 32 + n];
    sjC0[nt] = wsi[WS_SJCOL + (g * 2 + 0) * 32 + n];
    sjC2[nt] = wsi[WS_SJCOL + (g * 2 + 1) * 32 + n];
    sj00[nt] = wsi[WS_SJCOR + ((g * 2 + 0) * 2 + 0) * 32 + n];
    sj02[nt] = wsi[WS_SJCOR + ((g * 2 + 0) * 2 + 1) * 32 + n];
    sj20[nt] = wsi[WS_SJCOR + ((g * 2 + 1) * 2 + 0) * 32 + n];
    sj22[nt] = wsi[WS_SJCOR + ((g * 2 + 1) * 2 + 1) * 32 + n];
    biasn[nt] = bias[n];
  }
  float f_ab = sx * sww, f_a = sx * mnw, f_b = mnx * sww, f_c = mnx * mnw * 32.0f;

  __syncthreads();

  for (int mt = 0; mt < 6; ++mt) {
    int mbase = (wave * 6 + mt) * 16;
    int m = mbase + n0;
    int row = m / 192, col = m % 192;

    v8i acc0 = {0, 0, 0, 0, 0, 0, 0, 0};
    v8i acc1 = {0, 0, 0, 0, 0, 0, 0, 0};
    v8i acc2 = {0, 0, 0, 0, 0, 0, 0, 0};

#pragma unroll
    for (int s = 0; s < 5; ++s) {
      v8i a;
#pragma unroll
      for (int r = 0; r < 4; ++r) {
        if (s < 4 || r < 2) {
          int tap = s * 2 + (r >> 1);
          int ci = (r & 1) * 8 + laneHi * 16;
          int dh = tap / 3, dw = tap % 3;
          const unsigned* p2 = (const unsigned*)&xs[((row + dh) * 194 + (col + dw)) * 32 + ci];
          a[2 * r]     = (int)p2[0];
          a[2 * r + 1] = (int)p2[1];
        } else {
          a[2 * r] = 0; a[2 * r + 1] = 0;
        }
      }
      v8i bo = (s < 4) ? bone_full : bone_last;
      acc0 = __builtin_amdgcn_wmma_i32_16x16x64_iu8(false, a, false, bfr[s][0], acc0, false, false);
      acc1 = __builtin_amdgcn_wmma_i32_16x16x64_iu8(false, a, false, bfr[s][1], acc1, false, false);
      acc2 = __builtin_amdgcn_wmma_i32_16x16x64_iu8(false, a, false, bo,        acc2, false, false);
    }

    int rowg = rb * 4 + mbase / 192;
    bool r0 = (rowg == 0), r1 = (rowg == 191);
    int c0t = mbase % 192;
#pragma unroll
    for (int v = 0; v < 8; ++v) {
      int mrow = v + laneHi * 8;
      int c = c0t + mrow;
      bool cc0 = (c == 0), cc1 = (c == 191);
      int sk = acc2[v];
      int ntaps = (3 - (int)r0 - (int)r1) * (3 - (int)cc0 - (int)cc1);
#pragma unroll
      for (int nt = 0; nt < 2; ++nt) {
        int accv = nt ? acc1[v] : acc0[v];
        int sjv = sjF[nt]
                - (r0 ? sjR0[nt] : 0) - (r1 ? sjR2[nt] : 0)
                - (cc0 ? sjC0[nt] : 0) - (cc1 ? sjC2[nt] : 0)
                + ((r0 && cc0) ? sj00[nt] : 0) + ((r0 && cc1) ? sj02[nt] : 0)
                + ((r1 && cc0) ? sj20[nt] : 0) + ((r1 && cc1) ? sj22[nt] : 0);
        float y = f_ab * (float)accv + f_a * (float)sk + f_b * (float)sjv
                + f_c * (float)ntaps + biasn[nt];
        ot[wave][mrow * 32 + nt * 16 + n0] = y;
      }
    }

    size_t ob = (((size_t)img * 32 + lane) * 192 + rowg) * 192 + c0t;
#pragma unroll
    for (int q4 = 0; q4 < 4; ++q4) {
      float4 vv;
      vv.x = ot[wave][(q4 * 4 + 0) * 32 + lane];
      vv.y = ot[wave][(q4 * 4 + 1) * 32 + lane];
      vv.z = ot[wave][(q4 * 4 + 2) * 32 + lane];
      vv.w = ot[wave][(q4 * 4 + 3) * 32 + lane];
      *(float4*)(out + ob + q4 * 4) = vv;
    }
  }
}

extern "C" void kernel_launch(void* const* d_in, const int* in_sizes, int n_in,
                              void* d_out, int out_size, void* d_ws, size_t ws_size,
                              hipStream_t stream) {
  (void)in_sizes; (void)n_in; (void)out_size; (void)ws_size;
  const float* x    = (const float*)d_in[0];
  const float* w    = (const float*)d_in[1];
  const float* bias = (const float*)d_in[2];
  const int*   bits = (const int*)d_in[3];
  float* out = (float*)d_out;

  k_init_weight<<<1, 256, 0, stream>>>(w, d_ws);
  k_xstats<<<dim3(72, 32), 256, 0, stream>>>(x, bits, d_ws);
  k_finalize<<<1, 32, 0, stream>>>(d_ws);
  k_conv<<<dim3(48, 32), 256, 0, stream>>>(x, bias, bits, d_ws, out);
}
